// MoireAttention_18081812316395
// MI455X (gfx1250) — matrix operand order repaired and checked by reading, not compiled, not run on hardware
//
#include <hip/hip_runtime.h>
#include <hip/hip_bf16.h>
#include <cstddef>

// ---------------- CDNA5 WMMA types/helpers ----------------
typedef __attribute__((ext_vector_type(16))) __bf16 v16bf;
typedef __attribute__((ext_vector_type(8)))  __bf16 v8bf;
typedef __attribute__((ext_vector_type(8)))  float  v8f;

__device__ inline unsigned short f2bf(float f) {
    unsigned int u = __float_as_uint(f);
    unsigned int r = u + 0x7FFFu + ((u >> 16) & 1u);   // round-to-nearest-even
    return (unsigned short)(r >> 16);
}

__device__ inline v16bf mk16(v8bf lo, v8bf hi) {
    v16bf r;
#pragma unroll
    for (int i = 0; i < 8; i++) { r[i] = lo[i]; r[i + 8] = hi[i]; }
    return r;
}

// A fragment 16x32 (MxK), bf16, from row-major [row][k] buffer (stride in elems).
// lane<16: row=lane, K{0..7,16..23}; lane>=16: row=lane-16, K{8..15,24..31}.
__device__ inline v16bf load_afrag(const unsigned short* base, int stride, int lane) {
    int row = lane & 15;
    int k0  = (lane >> 4) ? 8 : 0;
    const v8bf* p0 = (const v8bf*)(base + (size_t)row * stride + k0);
    const v8bf* p1 = (const v8bf*)(base + (size_t)row * stride + k0 + 16);
    return mk16(*p0, *p1);
}

// B fragment 32x16 (KxN), bf16, from [n][k] buffer (stride in elems).
// lane<16: col n=lane, K{0..7,16..23}; lane>=16: col n=lane-16, K{8..15,24..31}.
__device__ inline v16bf load_bfrag(const unsigned short* base, int stride, int lane) {
    int n  = lane & 15;
    int k0 = (lane >> 4) ? 8 : 0;
    const v8bf* p0 = (const v8bf*)(base + (size_t)n * stride + k0);
    const v8bf* p1 = (const v8bf*)(base + (size_t)n * stride + k0 + 16);
    return mk16(*p0, *p1);
}

__device__ inline v8f wmma_bf16(v16bf a, v16bf b, v8f c) {
    return __builtin_amdgcn_wmma_f32_16x16x32_bf16(false, a, false, b, (short)0, c,
                                                   false, false);
}

// ---------------- Kernel 1: QKV projection (bf16 WMMA GEMM) ----------------
// out[8192,768] = x[8192,256] @ W[256,768] + b ; split to Q/K/V bf16 [B][H][N][32]
#define MT   128
#define NT   64
#define KTS  32
#define APAD 40   // padded k-stride (multiple of 8 -> 16B aligned frag loads)

__global__ __launch_bounds__(256) void qkv_gemm(
    const float* __restrict__ x, const float* __restrict__ W,
    const float* __restrict__ bias,
    unsigned short* __restrict__ Qb, unsigned short* __restrict__ Kb,
    unsigned short* __restrict__ Vb)
{
    __shared__ unsigned short As[MT][APAD];
    __shared__ unsigned short Bs[NT][APAD];

    const int tid  = threadIdx.x;
    const int lane = tid & 31;
    const int w    = tid >> 5;        // 8 waves
    const int wy   = w >> 1;          // 4 x 2 wave grid, 32x32 per wave
    const int wx   = w & 1;
    const int M0   = blockIdx.y * MT;
    const int N0   = blockIdx.x * NT;

    v8f c00 = {}, c01 = {}, c10 = {}, c11 = {};

    for (int kt = 0; kt < 256; kt += KTS) {
        // Stage A: x tile 128x32 (f32 -> bf16)
        {
            int r  = tid >> 3;
            int kg = (tid & 7) * 4;
#pragma unroll
            for (int ch = 0; ch < 4; ch++) {
                int row = r + ch * 32;
                const float4 v = *(const float4*)(x + (size_t)(M0 + row) * 256 + kt + kg);
                unsigned short* d = &As[row][kg];
                d[0] = f2bf(v.x); d[1] = f2bf(v.y); d[2] = f2bf(v.z); d[3] = f2bf(v.w);
            }
        }
        // Stage B transposed: W[kt..kt+31][N0..N0+63] -> Bs[n][k]
        {
            int n = tid & 63;
            int kb = tid >> 6;
#pragma unroll
            for (int ch = 0; ch < 8; ch++) {
                int k = kb + ch * 4;
                Bs[n][k] = f2bf(W[(size_t)(kt + k) * 768 + N0 + n]);
            }
        }
        __syncthreads();

        v16bf a0 = load_afrag(&As[wy * 32][0],      APAD, lane);
        v16bf a1 = load_afrag(&As[wy * 32 + 16][0], APAD, lane);
        v16bf b0 = load_bfrag(&Bs[wx * 32][0],      APAD, lane);
        v16bf b1 = load_bfrag(&Bs[wx * 32 + 16][0], APAD, lane);
        c00 = wmma_bf16(a0, b0, c00);
        c01 = wmma_bf16(a0, b1, c01);
        c10 = wmma_bf16(a1, b0, c10);
        c11 = wmma_bf16(a1, b1, c11);
        __syncthreads();
    }

    const int half = lane >> 4, col0 = lane & 15;
    auto store_tile = [&](const v8f& c, int mi, int ni) {
#pragma unroll
        for (int r = 0; r < 8; r++) {
            int grow = M0 + wy * 32 + mi + r + 8 * half;
            int gcol = N0 + wx * 32 + ni + col0;
            float v = c[r] + bias[gcol];
            int s    = gcol >> 8;          // 0=Q 1=K 2=V
            int h    = (gcol >> 5) & 7;
            int dim  = gcol & 31;
            int bidx = grow >> 10;
            int nidx = grow & 1023;
            size_t off = ((((size_t)bidx * 8 + h) * 1024) + nidx) * 32 + dim;
            unsigned short bv = f2bf(v);
            if (s == 0) Qb[off] = bv;
            else if (s == 1) Kb[off] = bv;
            else Vb[off] = bv;
        }
    };
    store_tile(c00, 0, 0);  store_tile(c01, 0, 16);
    store_tile(c10, 16, 0); store_tile(c11, 16, 16);
}

// ---------------- Kernel 2: fused moire flash-attention ----------------
// One workgroup: 8 waves = 8 heads, one 32-row query block of one batch.
// Each wave: 2 Q fragments, 4 f32 O accum tiles; online softmax over 32-key
// tiles; moire exp/log pair folded to a clamped quadratic.
#define KP 40

__global__ __launch_bounds__(256) void moire_attn(
    const unsigned short* __restrict__ Qb, const unsigned short* __restrict__ Kb,
    const unsigned short* __restrict__ Vb, const float* __restrict__ adj,
    const unsigned char* __restrict__ mask, const float* __restrict__ shifts,
    const float* __restrict__ widths, const float* __restrict__ slw,
    float* __restrict__ out)
{
    __shared__ unsigned short Ks[8][32][KP];   // per-head K tile [key][dim]
    __shared__ unsigned short Vts[8][32][KP];  // per-head V tile transposed [dim][key]
    __shared__ unsigned short Ps[8][32][32];   // per-head P tile (C->A relayout)
    __shared__ float maskf[1024];

    const int tid  = threadIdx.x;
    const int lane = tid & 31;
    const int h    = tid >> 5;                 // wave == head
    const int b    = blockIdx.x >> 5;          // 32 q-blocks per batch
    const int q0   = (blockIdx.x & 31) << 5;   // 32 query rows per workgroup

    for (int i = tid; i < 1024; i += 256)
        maskf[i] = mask[(size_t)b * 1024 + i] ? 1.0f : 0.0f;

    const float sh     = shifts[h];
    const float invw   = 1.0f / fmaxf(widths[h], 0.5f);
    const float selfw  = slw[h];
    const float scale  = 0.17677669529663687f;  // 1/sqrt(32)
    const float LOGEPS = -13.815510557964274f;  // ln(1e-6)

    // Two Q fragments (rows q0..q0+15 and q0+16..q0+31), straight from global bf16
    const size_t qoff = (((size_t)b * 8 + h) * 1024 + q0) * 32;
    v16bf aq0 = load_afrag(Qb + qoff,           32, lane);
    v16bf aq1 = load_afrag(Qb + qoff + 16 * 32, 32, lane);

    const int half = lane >> 4, ln = lane & 15;
    float mrow0[8], lrow0[8], mrow1[8], lrow1[8];
    v8f o00 = {}, o01 = {}, o10 = {}, o11 = {};
#pragma unroll
    for (int r = 0; r < 8; r++) {
        mrow0[r] = -1e30f; lrow0[r] = 0.0f;
        mrow1[r] = -1e30f; lrow1[r] = 0.0f;
    }

    __syncthreads();

    // online-softmax update for one 16-row block against the current 32 keys
    auto smax = [&](const v8f& sa, const v8f& sb, float (&mr)[8], float (&lr)[8],
                    v8f& oa, v8f& ob, int qb, int pr, int kt) {
#pragma unroll
        for (int r = 0; r < 8; r++) {
            const int q = qb + r + 8 * half;
            const float mqv = maskf[q];
            const size_t abase = ((size_t)b * 1024 + q) * 1024 + kt;

            const int k0i = kt + ln;
            float a0v = adj[abase + ln];
            float t0  = a0v - sh;
            float v0  = sa[r] * scale + fmaxf(-t0 * t0 * invw, LOGEPS);
            if (q == k0i) v0 += selfw;
            v0 = (mqv * maskf[k0i] != 0.0f) ? v0 : -1000000.0f;

            const int k1i = kt + 16 + ln;
            float a1v = adj[abase + 16 + ln];
            float t1  = a1v - sh;
            float v1  = sb[r] * scale + fmaxf(-t1 * t1 * invw, LOGEPS);
            if (q == k1i) v1 += selfw;
            v1 = (mqv * maskf[k1i] != 0.0f) ? v1 : -1000000.0f;

            // row max across 16 lanes (wave32: xor 1..8 stays in half-wave)
            float tm = fmaxf(v0, v1);
            tm = fmaxf(tm, __shfl_xor(tm, 1));
            tm = fmaxf(tm, __shfl_xor(tm, 2));
            tm = fmaxf(tm, __shfl_xor(tm, 4));
            tm = fmaxf(tm, __shfl_xor(tm, 8));
            float mnew  = fmaxf(mr[r], tm);
            float alpha = __expf(mr[r] - mnew);
            float e0 = __expf(v0 - mnew);
            float e1 = __expf(v1 - mnew);
            float rs = e0 + e1;
            rs += __shfl_xor(rs, 1);
            rs += __shfl_xor(rs, 2);
            rs += __shfl_xor(rs, 4);
            rs += __shfl_xor(rs, 8);
            lr[r] = lr[r] * alpha + rs;
            mr[r] = mnew;
            oa[r] *= alpha;
            ob[r] *= alpha;

            Ps[h][pr + r + 8 * half][ln]      = f2bf(e0);
            Ps[h][pr + r + 8 * half][16 + ln] = f2bf(e1);
        }
    };

    for (int kt = 0; kt < 1024; kt += 32) {
        // Stage K (plain) and V (transposed) tiles for this head
        {
            const size_t koff = (((size_t)b * 8 + h) * 1024 + kt + lane) * 32;
            const uint4* ks = (const uint4*)(Kb + koff);
            uint4* kd = (uint4*)&Ks[h][lane][0];
            kd[0] = ks[0]; kd[1] = ks[1]; kd[2] = ks[2]; kd[3] = ks[3];
            const uint4* vs = (const uint4*)(Vb + koff);
            union { uint4 q[4]; unsigned short u[32]; } vrow;
            vrow.q[0] = vs[0]; vrow.q[1] = vs[1]; vrow.q[2] = vs[2]; vrow.q[3] = vs[3];
#pragma unroll
            for (int d = 0; d < 32; d++) Vts[h][d][lane] = vrow.u[d];
        }
        // warm the near cache for the next tile's adj rows (global_prefetch_b8);
        // lane l covers query row q0+l, one line spans the 32-float tile row
        if (kt + 32 < 1024) {
            const size_t pbase = ((size_t)b * 1024 + q0 + lane) * 1024 + kt + 32;
            __builtin_prefetch(adj + pbase, 0, 1);
            __builtin_prefetch(adj + pbase + 16, 0, 1);
        }
        __syncthreads();

        // S = Q x K^T  (four 16x16 tiles: 2 row-blocks x 2 key-halves)
        v16bf bk0 = load_bfrag(&Ks[h][0][0],  KP, lane);
        v16bf bk1 = load_bfrag(&Ks[h][16][0], KP, lane);
        v8f z = {};
        v8f s00 = wmma_bf16(aq0, bk0, z);
        v8f s01 = wmma_bf16(aq0, bk1, z);
        v8f s10 = wmma_bf16(aq1, bk0, z);
        v8f s11 = wmma_bf16(aq1, bk1, z);

        smax(s00, s01, mrow0, lrow0, o00, o01, q0,      0,  kt);
        smax(s10, s11, mrow1, lrow1, o10, o11, q0 + 16, 16, kt);
        __syncthreads();

        // O += P x V  (P relayout C->A via LDS; V pre-transposed [dim][key])
        v16bf ap0 = load_afrag(&Ps[h][0][0],  32, lane);
        v16bf ap1 = load_afrag(&Ps[h][16][0], 32, lane);
        v16bf bv0 = load_bfrag(&Vts[h][0][0],  KP, lane);
        v16bf bv1 = load_bfrag(&Vts[h][16][0], KP, lane);
        o00 = wmma_bf16(ap0, bv0, o00);
        o01 = wmma_bf16(ap0, bv1, o01);
        o10 = wmma_bf16(ap1, bv0, o10);
        o11 = wmma_bf16(ap1, bv1, o11);
        __syncthreads();
    }

    // out[b][q][h*32+dim] = O / l
#pragma unroll
    for (int r = 0; r < 8; r++) {
        int qA = q0 + r + 8 * half;
        int qB = qA + 16;
        float ia = 1.0f / lrow0[r];
        float ib = 1.0f / lrow1[r];
        size_t offA = ((size_t)b * 1024 + qA) * 256 + h * 32;
        size_t offB = ((size_t)b * 1024 + qB) * 256 + h * 32;
        out[offA + ln]      = o00[r] * ia;
        out[offA + 16 + ln] = o01[r] * ia;
        out[offB + ln]      = o10[r] * ib;
        out[offB + 16 + ln] = o11[r] * ib;
    }
}

// ---------------- launch ----------------
extern "C" void kernel_launch(void* const* d_in, const int* in_sizes, int n_in,
                              void* d_out, int out_size, void* d_ws, size_t ws_size,
                              hipStream_t stream) {
    (void)in_sizes; (void)n_in; (void)out_size; (void)ws_size;
    const float*         x      = (const float*)d_in[0];
    const float*         adj    = (const float*)d_in[1];
    const unsigned char* mask   = (const unsigned char*)d_in[2]; // bool array
    const float*         W_qkv  = (const float*)d_in[3];
    const float*         b_qkv  = (const float*)d_in[4];
    const float*         shifts = (const float*)d_in[5];
    const float*         widths = (const float*)d_in[6];
    const float*         selfw  = (const float*)d_in[7];
    float*               out    = (float*)d_out;

    // workspace: Q/K/V bf16 [8][8][1024][32] = 2M elems each (4 MB each)
    const size_t QKV_ELEMS = (size_t)8 * 8 * 1024 * 32;
    unsigned short* Qb = (unsigned short*)d_ws;
    unsigned short* Kb = Qb + QKV_ELEMS;
    unsigned short* Vb = Kb + QKV_ELEMS;

    dim3 g1(768 / NT, 8192 / MT);     // 12 x 64
    qkv_gemm<<<g1, 256, 0, stream>>>(x, W_qkv, b_qkv, Qb, Kb, Vb);

    moire_attn<<<dim3(8 * 32), 256, 0, stream>>>(Qb, Kb, Vb, adj, mask,
                                                 shifts, widths, selfw, out);
}
